// GATv2NodeClassifier_31920196944057
// MI455X (gfx1250) — matrix operand order repaired and checked by reading, not compiled, not run on hardware
//
#include <hip/hip_runtime.h>
#include <hip/hip_bf16.h>
#include <stdint.h>

typedef __attribute__((ext_vector_type(16))) _Float16 v16h;
typedef __attribute__((ext_vector_type(8)))  _Float16 v8h;
typedef __attribute__((ext_vector_type(8)))  float    v8f;

// ---------------------------------------------------------------------------
// Elementwise f32 -> f16
// ---------------------------------------------------------------------------
__global__ __launch_bounds__(256)
void f32_to_f16_kernel(const float* __restrict__ in, _Float16* __restrict__ out, long long n) {
  long long i = (long long)blockIdx.x * blockDim.x + threadIdx.x;
  if (i < n) out[i] = (_Float16)in[i];
}

// W [K x M] f32 (row-major) -> Wt [Mpad x K] f16 (transposed, zero-padded)
__global__ __launch_bounds__(256)
void w_transpose_f16_kernel(const float* __restrict__ W, _Float16* __restrict__ Wt,
                            int K, int M, int Mpad) {
  int i = blockIdx.x * blockDim.x + threadIdx.x;
  if (i >= Mpad * K) return;
  int m = i / K, k = i - m * K;
  Wt[i] = (m < M) ? (_Float16)W[(size_t)k * M + m] : (_Float16)0.0f;
}

// ---------------------------------------------------------------------------
// WMMA GEMM: Y[N x M] = X[N x K](f16) * W[K x M] + bias, W given as Wt[MT*16 x K] f16.
// One wave per 16-row tile, MT column tiles held in v8f accumulators.
// A fragment (16x32 f16, wave32): lane<16 -> row=lane&15, halves K{0..7},{16..23};
//                                 lane>=16 ->            halves K{8..15},{24..31}.
// B fragment (32x16 f16): lane n = lane&15, same K halves as A: K{0..7},{16..23} (lanes 0-15) /
//                         K{8..15},{24..31} (lanes 16-31)  -> two ds_load_b128 from Wt rows.
// ---------------------------------------------------------------------------
template<int MT>
__global__ __launch_bounds__(256)
void gemm_wmma_f16(const _Float16* __restrict__ X, const _Float16* __restrict__ Wt,
                   const float* __restrict__ bias, float* __restrict__ Y,
                   int Nrows, int K, int M) {
  __shared__ _Float16 lw[8192];                    // up to 64x128 f16 = 16 KB
  const int total = MT * 16 * K;
  for (int i = threadIdx.x * 8; i < total; i += blockDim.x * 8)
    *(v8h*)(lw + i) = *(const v8h*)(Wt + i);
  __syncthreads();

  const int wave = threadIdx.x >> 5;
  const int lane = threadIdx.x & 31;
  const int half = lane >> 4;                      // 0: lanes 0-15, 1: lanes 16-31
  const int l16  = lane & 15;
  const int rowBase = (blockIdx.x * 8 + wave) * 16;
  if (rowBase >= Nrows) return;                    // wave-uniform; EXEC stays all-ones

  v8f acc[MT] = {};

  int arow = rowBase + l16;
  if (arow >= Nrows) arow = Nrows - 1;             // clamp (N is a multiple of 16 anyway)
  const _Float16* ap = X + (size_t)arow * K;

  for (int k0 = 0; k0 < K; k0 += 32) {
    v8h a0 = *(const v8h*)(ap + k0 + half * 8);
    v8h a1 = *(const v8h*)(ap + k0 + half * 8 + 16);
    v16h a;
#pragma unroll
    for (int i = 0; i < 8; ++i) { a[i] = a0[i]; a[i + 8] = a1[i]; }
#pragma unroll
    for (int t = 0; t < MT; ++t) {
      const _Float16* bp = lw + (size_t)(t * 16 + l16) * K + k0 + half * 8;
      v8h b0 = *(const v8h*)bp;
      v8h b1 = *(const v8h*)(bp + 16);
      v16h b;
#pragma unroll
      for (int i = 0; i < 8; ++i) { b[i] = b0[i]; b[i + 8] = b1[i]; }
      acc[t] = __builtin_amdgcn_wmma_f32_16x16x32_f16(
          /*neg_a=*/false, a, /*neg_b=*/false, b,
          /*c_mod=*/(short)0, acc[t], /*reuse_a=*/false, /*reuse_b=*/false);
    }
  }

#pragma unroll
  for (int t = 0; t < MT; ++t) {
#pragma unroll
    for (int i = 0; i < 8; ++i) {
      int r = rowBase + half * 8 + i;               // C/D layout: VGPR i -> M = i (+8)
      int c = t * 16 + l16;
      if (r < Nrows && c < M) Y[(size_t)r * M + c] = acc[t][i] + bias[c];
    }
  }
}

// ---------------------------------------------------------------------------
// Edge passes: segment-softmax + scatter (3-pass recompute, no per-edge storage)
// ---------------------------------------------------------------------------
__device__ __forceinline__ unsigned ford(float f) {         // monotonic float->uint
  unsigned u = __float_as_uint(f);
  return (u >> 31) ? ~u : (u | 0x80000000u);
}
__device__ __forceinline__ float fdec(unsigned e) {
  return __uint_as_float((e >> 31) ? (e ^ 0x80000000u) : ~e);
}

template<int H, int C>
__device__ __forceinline__ float edge_logit(const float* __restrict__ xls,
                                            const float* __restrict__ xrd,
                                            const float* __restrict__ atth) {
  float acc = 0.f;
#pragma unroll
  for (int c = 0; c < C; ++c) {
    float e = xls[c] + xrd[c];
    e = e > 0.f ? e : 0.2f * e;                     // leaky_relu(0.2)
    acc = fmaf(e, atth[c], acc);
  }
  return acc;
}

template<int H, int C>
__global__ __launch_bounds__(256)
void edge_pass_kernel(const long long* __restrict__ ei, long long E, int N,
                      const float* __restrict__ xl, const float* __restrict__ xr,
                      const float* __restrict__ att,
                      unsigned* __restrict__ menc, float* __restrict__ ssum,
                      float* __restrict__ out, int pass) {
  long long t = (long long)blockIdx.x * blockDim.x + threadIdx.x;
  long long Et = E + N;                             // self-loops appended
  if (t >= Et * H) return;
  int h = (int)(t % H);
  long long e = t / H;
  long long s = (e < E) ? ei[e]     : (e - E);
  long long d = (e < E) ? ei[E + e] : (e - E);
  const float* xls = xl + ((size_t)s * H + h) * C;
  const float* xrd = xr + ((size_t)d * H + h) * C;
  float lg = edge_logit<H, C>(xls, xrd, att + (size_t)h * C);
  size_t dh = (size_t)d * H + h;
  if (pass == 0) {
    atomicMax(&menc[dh], ford(lg));
  } else if (pass == 1) {
    atomicAdd(&ssum[dh], __expf(lg - fdec(menc[dh])));
  } else {
    float alpha = __expf(lg - fdec(menc[dh])) / ssum[dh];
    float* o = out + dh * C;
#pragma unroll
    for (int c = 0; c < C; ++c) atomicAdd(&o[c], xls[c] * alpha);
  }
}

// h = elu(acc + bias1), emitted as f16 for the layer-2 WMMA GEMM
__global__ __launch_bounds__(256)
void elu_bias_to_f16_kernel(const float* __restrict__ acc, const float* __restrict__ bias,
                            _Float16* __restrict__ h16, int n, int M) {
  int i = blockIdx.x * blockDim.x + threadIdx.x;
  if (i >= n) return;
  float v = acc[i] + bias[i % M];
  v = v > 0.f ? v : (__expf(v) - 1.f);
  h16[i] = (_Float16)v;
}

__global__ __launch_bounds__(256)
void add_bias_kernel(float* __restrict__ out, const float* __restrict__ bias, int n, int M) {
  int i = blockIdx.x * blockDim.x + threadIdx.x;
  if (i < n) out[i] += bias[i % M];
}

// ---------------------------------------------------------------------------
// Launch
// ---------------------------------------------------------------------------
extern "C" void kernel_launch(void* const* d_in, const int* in_sizes, int n_in,
                              void* d_out, int out_size, void* d_ws, size_t ws_size,
                              hipStream_t stream) {
  (void)n_in; (void)out_size; (void)ws_size;
  const int N = 50000, IN = 128, HID = 64, OUT = 40, H1 = 8, C1 = 8;
  const int M2PAD = 48;

  const float*     x     = (const float*)d_in[0];
  const long long* ei    = (const long long*)d_in[1];   // int64 edge_index [2 x E]
  const float*     Wl1   = (const float*)d_in[2];
  const float*     bl1   = (const float*)d_in[3];
  const float*     Wr1   = (const float*)d_in[4];
  const float*     br1   = (const float*)d_in[5];
  const float*     att1  = (const float*)d_in[6];
  const float*     bias1 = (const float*)d_in[7];
  const float*     Wl2   = (const float*)d_in[8];
  const float*     bl2   = (const float*)d_in[9];
  const float*     Wr2   = (const float*)d_in[10];
  const float*     br2   = (const float*)d_in[11];
  const float*     att2  = (const float*)d_in[12];
  const float*     bias2 = (const float*)d_in[13];
  const long long  E     = in_sizes[1] / 2;
  float* out = (float*)d_out;

  // ---- workspace carve (256B aligned) ----
  char* p = (char*)d_ws;
  auto carve = [&](size_t bytes) { char* r = p; p += (bytes + 255) & ~(size_t)255; return r; };
  _Float16* x16  = (_Float16*)carve((size_t)N * IN * 2);
  _Float16* h16  = (_Float16*)carve((size_t)N * HID * 2);
  _Float16* wl1t = (_Float16*)carve((size_t)HID * IN * 2);
  _Float16* wr1t = (_Float16*)carve((size_t)HID * IN * 2);
  _Float16* wl2t = (_Float16*)carve((size_t)M2PAD * HID * 2);
  _Float16* wr2t = (_Float16*)carve((size_t)M2PAD * HID * 2);
  float*    xl1  = (float*)carve((size_t)N * HID * 4);
  float*    xr1  = (float*)carve((size_t)N * HID * 4);
  float*    xl2  = (float*)carve((size_t)N * OUT * 4);
  float*    xr2  = (float*)carve((size_t)N * OUT * 4);
  float*    acc1 = (float*)carve((size_t)N * HID * 4);
  unsigned* m1   = (unsigned*)carve((size_t)N * H1 * 4);
  float*    s1   = (float*)carve((size_t)N * H1 * 4);
  unsigned* m2   = (unsigned*)carve((size_t)N * 4);
  float*    s2   = (float*)carve((size_t)N * 4);

  // ---- zero accumulators (graph-capture-safe) ----
  hipMemsetAsync(acc1, 0, (size_t)N * HID * 4, stream);
  hipMemsetAsync(m1,   0, (size_t)N * H1 * 4, stream);
  hipMemsetAsync(s1,   0, (size_t)N * H1 * 4, stream);
  hipMemsetAsync(m2,   0, (size_t)N * 4, stream);
  hipMemsetAsync(s2,   0, (size_t)N * 4, stream);
  hipMemsetAsync(out,  0, (size_t)N * OUT * 4, stream);

  const int B = 256;
  auto blks = [&](long long n) { return (unsigned)((n + B - 1) / B); };

  // ---- precision conversion / weight transpose ----
  f32_to_f16_kernel<<<blks((long long)N * IN), B, 0, stream>>>(x, x16, (long long)N * IN);
  w_transpose_f16_kernel<<<blks(HID * IN), B, 0, stream>>>(Wl1, wl1t, IN, HID, HID);
  w_transpose_f16_kernel<<<blks(HID * IN), B, 0, stream>>>(Wr1, wr1t, IN, HID, HID);
  w_transpose_f16_kernel<<<blks(M2PAD * HID), B, 0, stream>>>(Wl2, wl2t, HID, OUT, M2PAD);
  w_transpose_f16_kernel<<<blks(M2PAD * HID), B, 0, stream>>>(Wr2, wr2t, HID, OUT, M2PAD);

  // ---- layer-1 GEMMs (WMMA): xl1 = x@Wl1+bl1, xr1 = x@Wr1+br1 ----
  unsigned gemmBlocks = (unsigned)(((N + 15) / 16 + 7) / 8);
  gemm_wmma_f16<4><<<gemmBlocks, B, 0, stream>>>(x16, wl1t, bl1, xl1, N, IN, HID);
  gemm_wmma_f16<4><<<gemmBlocks, B, 0, stream>>>(x16, wr1t, br1, xr1, N, IN, HID);

  // ---- layer-1 edge passes: max -> sum -> weighted scatter into acc1 ----
  long long t1 = (E + N) * (long long)H1;
  edge_pass_kernel<H1, C1><<<blks(t1), B, 0, stream>>>(ei, E, N, xl1, xr1, att1, m1, s1, acc1, 0);
  edge_pass_kernel<H1, C1><<<blks(t1), B, 0, stream>>>(ei, E, N, xl1, xr1, att1, m1, s1, acc1, 1);
  edge_pass_kernel<H1, C1><<<blks(t1), B, 0, stream>>>(ei, E, N, xl1, xr1, att1, m1, s1, acc1, 2);

  // ---- h = elu(acc1 + bias1) -> f16 ----
  elu_bias_to_f16_kernel<<<blks((long long)N * HID), B, 0, stream>>>(acc1, bias1, h16, N * HID, HID);

  // ---- layer-2 GEMMs (WMMA, M=40 padded to 48) ----
  gemm_wmma_f16<3><<<gemmBlocks, B, 0, stream>>>(h16, wl2t, bl2, xl2, N, HID, OUT);
  gemm_wmma_f16<3><<<gemmBlocks, B, 0, stream>>>(h16, wr2t, br2, xr2, N, HID, OUT);

  // ---- layer-2 edge passes (H=1, C=40) scatter straight into d_out ----
  long long t2 = (E + N);
  edge_pass_kernel<1, OUT><<<blks(t2), B, 0, stream>>>(ei, E, N, xl2, xr2, att2, m2, s2, out, 0);
  edge_pass_kernel<1, OUT><<<blks(t2), B, 0, stream>>>(ei, E, N, xl2, xr2, att2, m2, s2, out, 1);
  edge_pass_kernel<1, OUT><<<blks(t2), B, 0, stream>>>(ei, E, N, xl2, xr2, att2, m2, s2, out, 2);

  // ---- mean over heads (H=1 -> identity) + bias2 ----
  add_bias_kernel<<<blks((long long)N * OUT), B, 0, stream>>>(out, bias2, N * OUT, OUT);
}
